// ImprovedFinGAT_42949673751
// MI455X (gfx1250) — matrix operand order repaired and checked by reading, not compiled, not run on hardware
//
#include <hip/hip_runtime.h>
#include <hip/hip_bf16.h>

typedef __attribute__((ext_vector_type(16))) _Float16 v16h;
typedef __attribute__((ext_vector_type(8)))  float    v8f;
typedef __attribute__((ext_vector_type(4)))  unsigned v4u;

#define HIDC 128
#define HEADSC 4

// ---------- helpers ----------
__device__ __forceinline__ unsigned f2mono(float f) {
  unsigned u = __float_as_uint(f);
  return (u & 0x80000000u) ? ~u : (u | 0x80000000u);
}
__device__ __forceinline__ float mono2f(unsigned u) {
  return (u & 0x80000000u) ? __uint_as_float(u & 0x7fffffffu) : __uint_as_float(~u);
}

__global__ void k_fill_f32(float* __restrict__ p, float v, long long cnt) {
  long long i = (long long)blockIdx.x * blockDim.x + threadIdx.x;
  if (i < cnt) p[i] = v;
}
__global__ void k_fill_u32(unsigned* __restrict__ p, unsigned v, long long cnt) {
  long long i = (long long)blockIdx.x * blockDim.x + threadIdx.x;
  if (i < cnt) p[i] = v;
}
// out[node*w + j] = bias[j]
__global__ void k_bcast(float* __restrict__ out, const float* __restrict__ bias,
                        long long cnt, int w) {
  long long i = (long long)blockIdx.x * blockDim.x + threadIdx.x;
  if (i < cnt) out[i] = bias[(int)(i % w)];
}

// ---------- pack f32 weight [K,N] row-major into WMMA B-fragment layout ----------
// layout: [kt][nt][lane(32)][elem(16)] ; lane<16: k=kt*32+elem, lane>=16: k=kt*32+16+elem
//         n = nt*16 + (lane&15)
__global__ void k_pack_b16(const float* __restrict__ W, _Float16* __restrict__ out,
                           int K, int N) {
  int tid = blockIdx.x * blockDim.x + threadIdx.x;
  if (tid >= K * N) return;
  int i    = tid & 15;
  int lane = (tid >> 4) & 31;
  int rest = tid >> 9;
  int ntiles = N >> 4;
  int nt = rest % ntiles;
  int kt = rest / ntiles;
  int k = kt * 32 + (lane < 16 ? i : 16 + i);
  int n = nt * 16 + (lane & 15);
  out[tid] = (_Float16)W[k * N + n];
}

// ---------- LayerNorm (+optional ELU/ReLU), wave-per-row, dual f32/f16 output ----------
// ACT: 0=none 1=elu 2=relu
template <int W, int ACT>
__global__ void k_ln_act(const float* __restrict__ in, const float* __restrict__ g,
                         const float* __restrict__ b, float* __restrict__ o32,
                         _Float16* __restrict__ o16, int n) {
  int wid  = (blockIdx.x * blockDim.x + threadIdx.x) >> 5;
  int lane = threadIdx.x & 31;
  if (wid >= n) return;
  const float* x = in + (size_t)wid * W;
  float v[W / 32];
  float s = 0.f, ss = 0.f;
#pragma unroll
  for (int j = 0; j < W / 32; ++j) {
    float t = x[lane + 32 * j];
    v[j] = t; s += t; ss += t * t;
  }
#pragma unroll
  for (int off = 16; off; off >>= 1) {
    s  += __shfl_xor(s, off, 32);
    ss += __shfl_xor(ss, off, 32);
  }
  float mu  = s * (1.0f / W);
  float var = ss * (1.0f / W) - mu * mu;
  float r   = rsqrtf(var + 1e-5f);
#pragma unroll
  for (int j = 0; j < W / 32; ++j) {
    int c = lane + 32 * j;
    float y = (v[j] - mu) * r * g[c] + b[c];
    if (ACT == 1) y = y > 0.f ? y : (expf(y) - 1.f);
    if (ACT == 2) y = fmaxf(y, 0.f);
    if (o32) o32[(size_t)wid * W + c] = y;
    if (o16) o16[(size_t)wid * W + c] = (_Float16)y;
  }
}

// ---------- WMMA GEMM: C[M,N] = act(A16[M,K] @ Bpacked[K,N] + bias) ----------
// one wave per 16x16 C tile; K multiple of 32, M/N multiples of 16.
__global__ void k_gemm_wmma(const _Float16* __restrict__ A,
                            const _Float16* __restrict__ Bp,
                            const float* __restrict__ bias,
                            float* __restrict__ C32, _Float16* __restrict__ C16,
                            int M, int N, int K, int relu) {
  int wid  = (blockIdx.x * blockDim.x + threadIdx.x) >> 5;
  int lane = threadIdx.x & 31;
  int ntiles = N >> 4;
  int tiles  = (M >> 4) * ntiles;
  if (wid >= tiles) return;
  int mt = wid / ntiles;
  int nt = wid - mt * ntiles;

  const _Float16* arow = A + (size_t)(mt * 16 + (lane & 15)) * K + (lane < 16 ? 0 : 16);
  const _Float16* bptr = Bp + (size_t)(nt * 32 + lane) * 16;
  const int bstep = ntiles * 512;  // halfs per k-tile

  v8f acc = {};
  for (int k0 = 0; k0 < K; k0 += 32) {
    union { v4u u[2]; v16h h; } a, b;
    a.u[0] = *(const v4u*)(arow + k0);
    a.u[1] = *(const v4u*)(arow + k0 + 8);
    b.u[0] = *(const v4u*)(bptr);
    b.u[1] = *(const v4u*)(bptr + 8);
    __builtin_prefetch(arow + k0 + 32, 0, 0);
    __builtin_prefetch(bptr + bstep, 0, 0);
    bptr += bstep;
    acc = __builtin_amdgcn_wmma_f32_16x16x32_f16(false, a.h, false, b.h,
                                                 (short)0, acc, false, false);
  }
  int col = nt * 16 + (lane & 15);
  float bv = bias ? bias[col] : 0.f;
  int mbase = mt * 16 + (lane < 16 ? 0 : 8);
#pragma unroll
  for (int r = 0; r < 8; ++r) {
    float yv = acc[r] + bv;
    if (relu) yv = fmaxf(yv, 0.f);
    size_t off = (size_t)(mbase + r) * N + col;
    if (C32) C32[off] = yv;
    if (C16) C16[off] = (_Float16)yv;
  }
}

// ---------- GATv2 edge passes (wave per (edge,head)) ----------
__global__ void k_edge_logits(const float* __restrict__ xl, const float* __restrict__ xr,
                              const float* __restrict__ att, const int* __restrict__ ei,
                              int E0, int ET, int H, int C,
                              float* __restrict__ logits, unsigned* __restrict__ mU) {
  int gw   = (blockIdx.x * blockDim.x + threadIdx.x) >> 5;
  int lane = threadIdx.x & 31;
  if (gw >= ET * H) return;
  int e = gw / H, h = gw - e * H;
  int s = e < E0 ? ei[e]      : (e - E0);
  int t = e < E0 ? ei[E0 + e] : (e - E0);
  int stride = H * C;
  const float* pl = xl + (size_t)s * stride + h * C;
  const float* pr = xr + (size_t)t * stride + h * C;
  const float* pa = att + h * C;
  float dot = 0.f;
  for (int j = lane; j < C; j += 32) {
    float v = pl[j] + pr[j];
    v = v > 0.f ? v : 0.2f * v;  // leaky_relu(0.2)
    dot += pa[j] * v;
  }
#pragma unroll
  for (int off = 16; off; off >>= 1) dot += __shfl_xor(dot, off, 32);
  if (lane == 0) {
    logits[(size_t)e * H + h] = dot;
    atomicMax(mU + (size_t)t * H + h, f2mono(dot));
  }
}

__global__ void k_edge_exp(float* __restrict__ logits, const unsigned* __restrict__ mU,
                           const int* __restrict__ ei, int E0, int ET, int H,
                           float* __restrict__ S) {
  int i = blockIdx.x * blockDim.x + threadIdx.x;
  if (i >= ET * H) return;
  int e = i / H, h = i - e * H;
  int t = e < E0 ? ei[E0 + e] : (e - E0);
  float m  = mono2f(mU[(size_t)t * H + h]);
  float ev = expf(logits[i] - m);
  logits[i] = ev;
  atomicAdd(S + (size_t)t * H + h, ev);
}

__global__ void k_edge_scatter(const float* __restrict__ xl, const float* __restrict__ ev,
                               const float* __restrict__ S, const int* __restrict__ ei,
                               int E0, int ET, int H, int C, float* __restrict__ out) {
  int gw   = (blockIdx.x * blockDim.x + threadIdx.x) >> 5;
  int lane = threadIdx.x & 31;
  if (gw >= ET * H) return;
  int e = gw / H, h = gw - e * H;
  int s = e < E0 ? ei[e]      : (e - E0);
  int t = e < E0 ? ei[E0 + e] : (e - E0);
  float w = ev[(size_t)e * H + h] / (S[(size_t)t * H + h] + 1e-16f);
  int stride = H * C;
  const float* pl = xl  + (size_t)s * stride + h * C;
  float*       po = out + (size_t)t * stride + h * C;
  for (int j = lane; j < C; j += 32) atomicAdd(po + j, pl[j] * w);
}

// ---------- small per-node FC (block per node, row staged in LDS) ----------
// act: 0=none 1=relu 2=tanh*0.1 3=sigmoid
__global__ void k_fc(const float* __restrict__ in, int K, const float* __restrict__ W,
                     const float* __restrict__ bias, float* __restrict__ out, int M,
                     int act) {
  __shared__ float row[128];
  int node = blockIdx.x;
  for (int j = threadIdx.x; j < K; j += blockDim.x) row[j] = in[(size_t)node * K + j];
  __syncthreads();
  int t = threadIdx.x;
  if (t < M) {
    float s = bias[t];
    for (int k = 0; k < K; ++k) s += row[k] * W[k * M + t];
    if (act == 1) s = fmaxf(s, 0.f);
    else if (act == 2) s = tanhf(s) * 0.1f;
    else if (act == 3) s = 1.f / (1.f + expf(-s));
    out[(size_t)node * M + t] = s;
  }
}

// ---------- host ----------
extern "C" void kernel_launch(void* const* d_in, const int* in_sizes, int n_in,
                              void* d_out, int out_size, void* d_ws, size_t ws_size,
                              hipStream_t stream) {
  const float* x    = (const float*)d_in[0];
  const int*   ei   = (const int*)  d_in[1];
  const float* fn_g = (const float*)d_in[2];  const float* fn_b = (const float*)d_in[3];
  const float* W_in = (const float*)d_in[4];  const float* b_in = (const float*)d_in[5];
  const float* Wl1  = (const float*)d_in[6];  const float* bl1  = (const float*)d_in[7];
  const float* Wr1  = (const float*)d_in[8];  const float* br1  = (const float*)d_in[9];
  const float* att1 = (const float*)d_in[10]; const float* bg1  = (const float*)d_in[11];
  const float* Wl2  = (const float*)d_in[12]; const float* bl2  = (const float*)d_in[13];
  const float* Wr2  = (const float*)d_in[14]; const float* br2  = (const float*)d_in[15];
  const float* att2 = (const float*)d_in[16]; const float* bg2  = (const float*)d_in[17];
  const float* ln1g = (const float*)d_in[18]; const float* ln1b = (const float*)d_in[19];
  const float* ln2g = (const float*)d_in[20]; const float* ln2b = (const float*)d_in[21];
  const float* Wrf = (const float*)d_in[22]; const float* brf = (const float*)d_in[23];
  const float* rfg = (const float*)d_in[24]; const float* rfb = (const float*)d_in[25];
  const float* Wcf = (const float*)d_in[26]; const float* bcf = (const float*)d_in[27];
  const float* cfg = (const float*)d_in[28]; const float* cfb = (const float*)d_in[29];
  const float* Wkf = (const float*)d_in[30]; const float* bkf = (const float*)d_in[31];
  const float* kfg = (const float*)d_in[32]; const float* kfb = (const float*)d_in[33];
  const float* Wr1h = (const float*)d_in[34]; const float* br1h = (const float*)d_in[35];
  const float* Wr2h = (const float*)d_in[36]; const float* br2h = (const float*)d_in[37];
  const float* Wr3h = (const float*)d_in[38]; const float* br3h = (const float*)d_in[39];
  const float* Wc1h = (const float*)d_in[40]; const float* bc1h = (const float*)d_in[41];
  const float* Wc2h = (const float*)d_in[42]; const float* bc2h = (const float*)d_in[43];
  const float* Wk1h = (const float*)d_in[44]; const float* bk1h = (const float*)d_in[45];
  const float* Wk2h = (const float*)d_in[46]; const float* bk2h = (const float*)d_in[47];

  const int n  = in_sizes[0] / HIDC;   // 20000
  const int E0 = in_sizes[1] / 2;      // 320000
  const int ET = E0 + n;               // edges + self-loops
  const int H4 = HEADSC * HIDC;        // 512

  // ---- workspace carve (256B aligned) ----
  char* wp = (char*)d_ws;
  auto carve = [&](size_t bytes) -> char* {
    char* r = wp; wp += (bytes + 255) & ~(size_t)255; return r;
  };
  _Float16* A16a = (_Float16*)carve((size_t)n * H4 * 2);
  _Float16* A16b = (_Float16*)carve((size_t)n * H4 * 2);
  _Float16* pW_in = (_Float16*)carve(128 * 128 * 2);
  _Float16* pWl1  = (_Float16*)carve(128 * 512 * 2);
  _Float16* pWr1  = (_Float16*)carve(128 * 512 * 2);
  _Float16* pWl2  = (_Float16*)carve(512 * 128 * 2);
  _Float16* pWr2  = (_Float16*)carve(512 * 128 * 2);
  _Float16* pWrf  = (_Float16*)carve(128 * 128 * 2);
  _Float16* pWcf  = (_Float16*)carve(128 * 128 * 2);
  _Float16* pWkf  = (_Float16*)carve(128 * 128 * 2);
  float* F0 = (float*)carve((size_t)n * H4 * 4);    // xl1
  float* F1 = (float*)carve((size_t)n * H4 * 4);    // xr1 -> gat1 out
  float* F3 = (float*)carve((size_t)n * HIDC * 4);  // h1/xl2
  float* F4 = (float*)carve((size_t)n * HIDC * 4);  // xr2 -> gat2 out
  float* F6 = (float*)carve((size_t)n * HIDC * 4);  // feature buf
  float* EL = (float*)carve((size_t)ET * HEADSC * 4);
  unsigned* mU = (unsigned*)carve((size_t)n * HEADSC * 4);
  float* S   = (float*)carve((size_t)n * HEADSC * 4);
  float* T64 = (float*)carve((size_t)n * 64 * 4);
  float* T32 = (float*)carve((size_t)n * 32 * 4);

  float* out_ret  = (float*)d_out;
  float* out_mov  = (float*)d_out + (size_t)n;
  float* out_rank = (float*)d_out + (size_t)3 * n;
  float* out_emb  = (float*)d_out + (size_t)4 * n;

  auto pack = [&](const float* Wf, _Float16* dst, int K, int Nn) {
    int tot = K * Nn;
    k_pack_b16<<<(tot + 255) / 256, 256, 0, stream>>>(Wf, dst, K, Nn);
  };
  auto gemm = [&](const _Float16* A, const _Float16* Bp, const float* bias,
                  float* C32, _Float16* C16, int M, int Nn, int K, int relu) {
    int waves = (M / 16) * (Nn / 16);
    k_gemm_wmma<<<(waves + 7) / 8, 256, 0, stream>>>(A, Bp, bias, C32, C16, M, Nn, K, relu);
  };
  auto fc = [&](const float* in, int K, const float* W, const float* b, float* out,
                int M, int act) {
    k_fc<<<n, 64, 0, stream>>>(in, K, W, b, out, M, act);
  };

  // pack all GEMM weights to f16 fragments
  pack(W_in, pW_in, 128, 128);
  pack(Wl1, pWl1, 128, 512);  pack(Wr1, pWr1, 128, 512);
  pack(Wl2, pWl2, 512, 128);  pack(Wr2, pWr2, 512, 128);
  pack(Wrf, pWrf, 128, 128);  pack(Wcf, pWcf, 128, 128);  pack(Wkf, pWkf, 128, 128);

  const int lnThreads = n * 32;
  // h = LN(x) -> f16
  k_ln_act<128, 0><<<(lnThreads + 255) / 256, 256, 0, stream>>>(x, fn_g, fn_b, nullptr, A16a, n);
  // h = relu(h @ W_in + b_in) -> f16
  gemm(A16a, pW_in, b_in, nullptr, A16b, n, 128, 128, 1);
  // xl1/xr1
  gemm(A16b, pWl1, bl1, F0, nullptr, n, 512, 128, 0);
  gemm(A16b, pWr1, br1, F1, nullptr, n, 512, 128, 0);

  // ---- GAT layer 1 (H=4, C=128) ----
  {
    long long nh = (long long)n * HEADSC;
    k_fill_u32<<<(int)((nh + 255) / 256), 256, 0, stream>>>(mU, 0u, nh);
    k_fill_f32<<<(int)((nh + 255) / 256), 256, 0, stream>>>(S, 0.f, nh);
    int waves = ET * HEADSC;
    k_edge_logits<<<(waves * 32 + 255) / 256, 256, 0, stream>>>(F0, F1, att1, ei, E0, ET,
                                                                HEADSC, HIDC, EL, mU);
    k_edge_exp<<<(waves + 255) / 256, 256, 0, stream>>>(EL, mU, ei, E0, ET, HEADSC, S);
    long long cnt = (long long)n * H4;
    k_bcast<<<(int)((cnt + 255) / 256), 256, 0, stream>>>(F1, bg1, cnt, H4);  // out := bias
    k_edge_scatter<<<(waves * 32 + 255) / 256, 256, 0, stream>>>(F0, EL, S, ei, E0, ET,
                                                                 HEADSC, HIDC, F1);
  }
  // h = elu(LN1(gat1)) -> f16 (N x 512)
  k_ln_act<512, 1><<<(lnThreads + 255) / 256, 256, 0, stream>>>(F1, ln1g, ln1b, nullptr, A16a, n);
  // xl2/xr2
  gemm(A16a, pWl2, bl2, F3, nullptr, n, 128, 512, 0);
  gemm(A16a, pWr2, br2, F4, nullptr, n, 128, 512, 0);

  // ---- GAT layer 2 (H=1, C=128) ----
  {
    k_fill_u32<<<(n + 255) / 256, 256, 0, stream>>>(mU, 0u, n);
    k_fill_f32<<<(n + 255) / 256, 256, 0, stream>>>(S, 0.f, n);
    int waves = ET;
    k_edge_logits<<<(waves * 32 + 255) / 256, 256, 0, stream>>>(F3, F4, att2, ei, E0, ET,
                                                                1, HIDC, EL, mU);
    k_edge_exp<<<(waves + 255) / 256, 256, 0, stream>>>(EL, mU, ei, E0, ET, 1, S);
    long long cnt = (long long)n * HIDC;
    k_bcast<<<(int)((cnt + 255) / 256), 256, 0, stream>>>(F4, bg2, cnt, HIDC);
    k_edge_scatter<<<(waves * 32 + 255) / 256, 256, 0, stream>>>(F3, EL, S, ei, E0, ET,
                                                                 1, HIDC, F4);
  }
  // emb = elu(LN2(gat2)) -> f32 (into d_out) + f16 for feature GEMMs
  k_ln_act<128, 1><<<(lnThreads + 255) / 256, 256, 0, stream>>>(F4, ln2g, ln2b, out_emb, A16b, n);

  // ---- returns head ----
  gemm(A16b, pWrf, brf, F6, nullptr, n, 128, 128, 0);
  k_ln_act<128, 2><<<(lnThreads + 255) / 256, 256, 0, stream>>>(F6, rfg, rfb, F6, nullptr, n);
  fc(F6, 128, Wr1h, br1h, T64, 64, 1);
  fc(T64, 64, Wr2h, br2h, T32, 32, 1);
  fc(T32, 32, Wr3h, br3h, out_ret, 1, 2);   // tanh * 0.1

  // ---- movement head ----
  gemm(A16b, pWcf, bcf, F6, nullptr, n, 128, 128, 0);
  k_ln_act<128, 2><<<(lnThreads + 255) / 256, 256, 0, stream>>>(F6, cfg, cfb, F6, nullptr, n);
  fc(F6, 128, Wc1h, bc1h, T64, 64, 1);
  fc(T64, 64, Wc2h, bc2h, out_mov, 2, 0);

  // ---- ranking head ----
  gemm(A16b, pWkf, bkf, F6, nullptr, n, 128, 128, 0);
  k_ln_act<128, 2><<<(lnThreads + 255) / 256, 256, 0, stream>>>(F6, kfg, kfb, F6, nullptr, n);
  fc(F6, 128, Wk1h, bk1h, T64, 64, 1);
  fc(T64, 64, Wk2h, bk2h, out_rank, 1, 3);  // sigmoid

  (void)n_in; (void)out_size; (void)ws_size;
}
